// GCN_23751169147431
// MI455X (gfx1250) — matrix operand order repaired and checked by reading, not compiled, not run on hardware
//
#include <hip/hip_runtime.h>

// ---------------------------------------------------------------------------
// 2-layer GAT + GraphNorm for MI455X (gfx1250, wave32, WMMA bf16->f32)
// ---------------------------------------------------------------------------

typedef __attribute__((ext_vector_type(16))) __bf16 v16bf;
typedef __attribute__((ext_vector_type(8)))  float  v8f;

#define NN    10000
#define NE    160000
#define ETOT  170000   // NE + NN self-loops
#define FIN   256
#define NH1   4
#define F1    1024     // NH1 * FIN
#define F2    256

// ------------------------------ utilities ---------------------------------

__global__ void k_zero32(unsigned* __restrict__ p, int n) {
  int i = blockIdx.x * blockDim.x + threadIdx.x;
  if (i < n) p[i] = 0u;
}

__global__ void k_f32_to_bf16(const float* __restrict__ in,
                              __bf16* __restrict__ out, int n) {
  int i = blockIdx.x * blockDim.x + threadIdx.x;
  if (i < n) out[i] = (__bf16)in[i];
}

// weights: f32 [K][N] row-major  ->  bf16 [N][K] (transposed, fragment-ready)
__global__ void k_f32_to_bf16_T(const float* __restrict__ in,
                                __bf16* __restrict__ out, int K, int N) {
  int t = blockIdx.x * blockDim.x + threadIdx.x;
  if (t >= K * N) return;
  int k = t / N, n = t - k * N;
  out[(size_t)n * K + k] = (__bf16)in[t];
}

// monotonic float <-> uint mapping so atomicMax(u32) == float max
__device__ __forceinline__ unsigned fenc(float f) {
  int i = __float_as_int(f);
  return (i >= 0) ? (unsigned(i) | 0x80000000u) : ~unsigned(i);
}
__device__ __forceinline__ float fdec(unsigned u) {
  int i = (u & 0x80000000u) ? int(u & 0x7FFFFFFFu) : ~int(u);
  return __int_as_float(i);
}

// ------------------------- WMMA bf16 GEMM (C=f32) --------------------------
// A: [M,K] row-major bf16.  Bt: [N,K] row-major bf16 (B transposed).
// Block tile 64(M) x 256(N), K-step 32. 8 waves: 2x4, each wave 32x64 out.

#define BM 64
#define BN 256
#define BK 32
#define LP (BK + 8)   // LDS pitch in halves: 40 halves = 80B (16B multiple)

__global__ __launch_bounds__(256) void k_wmma_gemm_bf16(
    const __bf16* __restrict__ A,   // [M,K]
    const __bf16* __restrict__ Bt,  // [N,K]
    float* __restrict__ C,          // [M,N]
    int M, int K, int N)
{
  __shared__ __bf16 As[BM][LP];     // A tile, row-major
  __shared__ __bf16 Bs[BN][LP];     // B tile, K-contiguous per column

  const int tid  = threadIdx.x;
  const int lane = tid & 31;
  const int wv   = tid >> 5;
  const int wm   = wv >> 2;           // 0..1
  const int wn   = wv & 3;            // 0..3
  const int tileM = blockIdx.y * BM;
  const int tileN = blockIdx.x * BN;

  v8f acc[2][4] = {};

  for (int k0 = 0; k0 < K; k0 += BK) {
    { // stage A tile: 64x32 halves, one 16B chunk per thread
      int r  = tid >> 2;
      int cc = (tid & 3) << 3;
      float4 v = make_float4(0.f, 0.f, 0.f, 0.f);
      int gr = tileM + r;
      if (gr < M) v = *(const float4*)(A + (size_t)gr * K + k0 + cc);
      *(float4*)(&As[r][cc]) = v;
    }
    { // stage B tile: 256 cols x 32 halves, one col (4x16B) per thread
      const float4* g = (const float4*)(Bt + (size_t)(tileN + tid) * K + k0);
      float4* l = (float4*)(&Bs[tid][0]);
      l[0] = g[0]; l[1] = g[1]; l[2] = g[2]; l[3] = g[3];
    }
    __syncthreads();

    // A fragment (16x32): lane<16 -> K 0-7 & 16-23, lane>=16 -> 8-15 & 24-31
    v16bf af[2], bfr[4];
    const int arow = lane & 15;
    const int akb  = (lane >> 4) << 3;   // 0 or 8
#pragma unroll
    for (int mi = 0; mi < 2; ++mi) {
      int rr = wm * 32 + mi * 16 + arow;
      ((float4*)&af[mi])[0] = *(const float4*)(&As[rr][akb]);
      ((float4*)&af[mi])[1] = *(const float4*)(&As[rr][akb + 16]);
    }
    // B fragment (32x16): lane = N col; lanes 0-15 K 0-7 & 16-23, 16-31 K 8-15 & 24-31
    const int bkb = (lane >> 4) << 3;    // 0 or 8
#pragma unroll
    for (int ni = 0; ni < 4; ++ni) {
      int cc = wn * 64 + ni * 16 + (lane & 15);
      ((float4*)&bfr[ni])[0] = *(const float4*)(&Bs[cc][bkb]);
      ((float4*)&bfr[ni])[1] = *(const float4*)(&Bs[cc][bkb + 16]);
    }
#pragma unroll
    for (int mi = 0; mi < 2; ++mi)
#pragma unroll
      for (int ni = 0; ni < 4; ++ni)
        acc[mi][ni] = __builtin_amdgcn_wmma_f32_16x16x32_bf16(
            false, af[mi], false, bfr[ni], (short)0, acc[mi][ni], false, false);
    __syncthreads();
  }

  // C layout: VGPR r -> M = r (lanes 0-15) or 8+r (lanes 16-31); N = lane&15
  const int rbase   = (lane >> 4) << 3;
  const int colBase = tileN + wn * 64 + (lane & 15);
#pragma unroll
  for (int mi = 0; mi < 2; ++mi) {
    int m0 = tileM + wm * 32 + mi * 16 + rbase;
    float* p = C + (size_t)m0 * N + colBase;
    if (m0 + 7 < M) {           // fast path: whole 8-row strip in range
#pragma unroll
      for (int ni = 0; ni < 4; ++ni)
#pragma unroll
        for (int r = 0; r < 8; ++r)
          p[r * N + ni * 16] = acc[mi][ni][r];
    } else {
#pragma unroll
      for (int ni = 0; ni < 4; ++ni)
#pragma unroll
        for (int r = 0; r < 8; ++r)
          if (m0 + r < M) p[r * N + ni * 16] = acc[mi][ni][r];
    }
  }
}

// ---------------------- per-node attention logits --------------------------
// one wave per (node, head): al_s = h . a_src, al_d = h . a_dst

__global__ __launch_bounds__(256) void k_attn_scores(
    const float* __restrict__ h, const float* __restrict__ asrc,
    const float* __restrict__ adst, float* __restrict__ als,
    float* __restrict__ ald, int Nn, int Hh, int Cc)
{
  int wid  = blockIdx.x * 8 + (threadIdx.x >> 5);
  int lane = threadIdx.x & 31;
  if (wid >= Nn * Hh) return;
  int node = wid / Hh, hh = wid - node * Hh;
  const float* hp = h + (size_t)node * Hh * Cc + hh * Cc;
  const float* ap = asrc + hh * Cc;
  const float* bp = adst + hh * Cc;
  float s1 = 0.f, s2 = 0.f;
  for (int c = lane; c < Cc; c += 32) {
    float v = hp[c];
    s1 += v * ap[c];
    s2 += v * bp[c];
  }
#pragma unroll
  for (int o = 16; o > 0; o >>= 1) {
    s1 += __shfl_xor(s1, o, 32);
    s2 += __shfl_xor(s2, o, 32);
  }
  if (lane == 0) { als[wid] = s1; ald[wid] = s2; }
}

// --------------------------- edge-wise kernels -----------------------------
// edge id >= E  ==>  synthesized self-loop (src = dst = id - E)

__global__ void k_edge_scores(const int* __restrict__ src,
                              const int* __restrict__ dst,
                              const float* __restrict__ als,
                              const float* __restrict__ ald,
                              float* __restrict__ ebuf,
                              unsigned* __restrict__ menc,
                              int Etot, int E, int Hh)
{
  int t = blockIdx.x * blockDim.x + threadIdx.x;
  if (t >= Etot * Hh) return;
  int e  = t / Hh, hh = t - e * Hh;
  int s  = (e < E) ? src[e] : (e - E);
  int d  = (e < E) ? dst[e] : (e - E);
  float v = als[s * Hh + hh] + ald[d * Hh + hh];
  v = (v > 0.f) ? v : 0.2f * v;           // leaky_relu, slope 0.2
  ebuf[t] = v;
  atomicMax(menc + d * Hh + hh, fenc(v)); // segment max per (dst, head)
}

__global__ void k_edge_expsum(const int* __restrict__ dst,
                              float* __restrict__ ebuf,
                              const unsigned* __restrict__ menc,
                              float* __restrict__ denom,
                              int Etot, int E, int Hh)
{
  int t = blockIdx.x * blockDim.x + threadIdx.x;
  if (t >= Etot * Hh) return;
  int e = t / Hh, hh = t - e * Hh;
  int d = (e < E) ? dst[e] : (e - E);
  float m = fdec(menc[d * Hh + hh]);
  float x = __expf(ebuf[t] - m);
  ebuf[t] = x;
  atomicAdd(denom + d * Hh + hh, x);
}

// out[dst] += (ex/denom[dst]) * h[src]   (one block per edge; L2-resident)
__global__ __launch_bounds__(256) void k_edge_aggregate(
    const int* __restrict__ src, const int* __restrict__ dst,
    const float* __restrict__ hfeat, const float* __restrict__ ebuf,
    const float* __restrict__ denom, float* __restrict__ outf,
    int Etot, int E, int Hh, int Cc)
{
  int e = blockIdx.x;
  if (e >= Etot) return;
  int s = (e < E) ? src[e] : (e - E);
  int d = (e < E) ? dst[e] : (e - E);
  int F = Hh * Cc;
  const float4* hv = (const float4*)(hfeat + (size_t)s * F);
  float* ob = outf + (size_t)d * F;
  for (int i = threadIdx.x; i < (F >> 2); i += blockDim.x) {
    int hh = (i << 2) / Cc;
    float w = ebuf[e * Hh + hh] / denom[d * Hh + hh];
    float4 v = hv[i];
    float* op = ob + (i << 2);
    atomicAdd(op + 0, w * v.x);
    atomicAdd(op + 1, w * v.y);
    atomicAdd(op + 2, w * v.z);
    atomicAdd(op + 3, w * v.w);
  }
}

// ------------------------------ GraphNorm ----------------------------------
// pass 1: per-column sum / sumsq of (x + bias)

__global__ __launch_bounds__(256) void k_gn_stats(
    const float* __restrict__ x, const float* __restrict__ bias,
    float* __restrict__ colsum, float* __restrict__ colsq,
    int Nn, int F, int rowsPer)
{
  int col = blockIdx.x * 256 + threadIdx.x;
  int r0  = blockIdx.y * rowsPer;
  int r1  = r0 + rowsPer; if (r1 > Nn) r1 = Nn;
  float b = bias[col];
  float s = 0.f, q = 0.f;
  for (int r = r0; r < r1; ++r) {
    float v = x[(size_t)r * F + col] + b;
    s += v; q += v * v;
  }
  atomicAdd(colsum + col, s);
  atomicAdd(colsq  + col, q);
}

// pass 2: y = relu(w*(v - ms*mean)/sqrt(var+eps)+b), emit bf16 for GEMM2
__global__ void k_gn_norm_relu_bf16(
    const float* __restrict__ x, const float* __restrict__ bias,
    const float* __restrict__ colsum, const float* __restrict__ colsq,
    const float* __restrict__ w, const float* __restrict__ bb,
    const float* __restrict__ ms, __bf16* __restrict__ out, int Nn, int F)
{
  int t = blockIdx.x * blockDim.x + threadIdx.x;
  if (t >= Nn * F) return;
  int c = t % F;
  float invN = 1.0f / (float)Nn;
  float mean = colsum[c] * invN;
  float ex2  = colsq[c] * invN;
  float m    = ms[c];
  // var of (v - m*mean) = E[v^2] - 2m*mean^2 + m^2*mean^2
  float var  = ex2 - (2.f * m - m * m) * mean * mean;
  float v    = x[t] + bias[c];
  float y    = w[c] * (v - m * mean) * rsqrtf(var + 1e-5f) + bb[c];
  y = fmaxf(y, 0.f);
  out[t] = (__bf16)y;
}

__global__ void k_add_bias(const float* __restrict__ in,
                           const float* __restrict__ b,
                           float* __restrict__ out, int Nn, int F)
{
  int t = blockIdx.x * blockDim.x + threadIdx.x;
  if (t >= Nn * F) return;
  out[t] = in[t] + b[t % F];
}

// ------------------------------ orchestration ------------------------------

static inline char* wsalloc(char*& cur, size_t bytes) {
  char* p = cur;
  cur += (bytes + 255) & ~size_t(255);
  return p;
}

extern "C" void kernel_launch(void* const* d_in, const int* in_sizes, int n_in,
                              void* d_out, int out_size, void* d_ws, size_t ws_size,
                              hipStream_t stream)
{
  (void)in_sizes; (void)n_in; (void)out_size; (void)ws_size;

  const float* x     = (const float*)d_in[0];
  const int*   ei    = (const int*)  d_in[1];
  const float* W1    = (const float*)d_in[2];
  const float* asrc1 = (const float*)d_in[3];
  const float* adst1 = (const float*)d_in[4];
  const float* b1    = (const float*)d_in[5];
  const float* gnw   = (const float*)d_in[6];
  const float* gnb   = (const float*)d_in[7];
  const float* gnms  = (const float*)d_in[8];
  const float* W2    = (const float*)d_in[9];
  const float* asrc2 = (const float*)d_in[10];
  const float* adst2 = (const float*)d_in[11];
  const float* b2    = (const float*)d_in[12];
  const int* srcE = ei;
  const int* dstE = ei + NE;

  char* cur = (char*)d_ws;
  __bf16*   A1   = (__bf16*)  wsalloc(cur, (size_t)NN * FIN * 2);
  __bf16*   W1b  = (__bf16*)  wsalloc(cur, (size_t)FIN * F1 * 2);   // [F1][FIN]
  float*    h1   = (float*)   wsalloc(cur, (size_t)NN * F1 * 4);
  float*    out1 = (float*)   wsalloc(cur, (size_t)NN * F1 * 4);
  __bf16*   A2   = (__bf16*)  wsalloc(cur, (size_t)NN * F1 * 2);
  __bf16*   W2b  = (__bf16*)  wsalloc(cur, (size_t)F1 * F2 * 2);    // [F2][F1]
  float*    h2   = (float*)   wsalloc(cur, (size_t)NN * F2 * 4);
  float*    out2 = (float*)   wsalloc(cur, (size_t)NN * F2 * 4);
  float*    als1 = (float*)   wsalloc(cur, (size_t)NN * NH1 * 4);
  float*    ald1 = (float*)   wsalloc(cur, (size_t)NN * NH1 * 4);
  unsigned* m1   = (unsigned*)wsalloc(cur, (size_t)NN * NH1 * 4);
  float*    den1 = (float*)   wsalloc(cur, (size_t)NN * NH1 * 4);
  float*    eb1  = (float*)   wsalloc(cur, (size_t)ETOT * NH1 * 4);
  float*    als2 = (float*)   wsalloc(cur, (size_t)NN * 4);
  float*    ald2 = (float*)   wsalloc(cur, (size_t)NN * 4);
  unsigned* m2   = (unsigned*)wsalloc(cur, (size_t)NN * 4);
  float*    den2 = (float*)   wsalloc(cur, (size_t)NN * 4);
  float*    eb2  = (float*)   wsalloc(cur, (size_t)ETOT * 4);
  float*    csum = (float*)   wsalloc(cur, (size_t)F1 * 4);
  float*    csq  = (float*)   wsalloc(cur, (size_t)F1 * 4);

  auto blks = [](int n) { return (n + 255) / 256; };

  // ---------------- layer 1: h1 = x @ W1 (bf16 WMMA) ----------------
  k_f32_to_bf16<<<blks(NN * FIN), 256, 0, stream>>>(x, A1, NN * FIN);
  k_f32_to_bf16_T<<<blks(FIN * F1), 256, 0, stream>>>(W1, W1b, FIN, F1);
  k_wmma_gemm_bf16<<<dim3(F1 / BN, (NN + BM - 1) / BM), 256, 0, stream>>>(
      A1, W1b, h1, NN, FIN, F1);

  // attention logits + segment softmax + aggregation
  k_attn_scores<<<(NN * NH1 + 7) / 8, 256, 0, stream>>>(
      h1, asrc1, adst1, als1, ald1, NN, NH1, FIN);
  k_zero32<<<blks(NN * NH1), 256, 0, stream>>>(m1, NN * NH1);
  k_zero32<<<blks(NN * NH1), 256, 0, stream>>>((unsigned*)den1, NN * NH1);
  k_zero32<<<blks(NN * F1), 256, 0, stream>>>((unsigned*)out1, NN * F1);
  k_edge_scores<<<blks(ETOT * NH1), 256, 0, stream>>>(
      srcE, dstE, als1, ald1, eb1, m1, ETOT, NE, NH1);
  k_edge_expsum<<<blks(ETOT * NH1), 256, 0, stream>>>(
      dstE, eb1, m1, den1, ETOT, NE, NH1);
  k_edge_aggregate<<<ETOT, 256, 0, stream>>>(
      srcE, dstE, h1, eb1, den1, out1, ETOT, NE, NH1, FIN);

  // GraphNorm(out1 + b1) -> relu -> bf16 A2
  k_zero32<<<blks(F1), 256, 0, stream>>>((unsigned*)csum, F1);
  k_zero32<<<blks(F1), 256, 0, stream>>>((unsigned*)csq, F1);
  k_gn_stats<<<dim3(F1 / 256, 64), 256, 0, stream>>>(
      out1, b1, csum, csq, NN, F1, (NN + 63) / 64);
  k_gn_norm_relu_bf16<<<blks(NN * F1), 256, 0, stream>>>(
      out1, b1, csum, csq, gnw, gnb, gnms, A2, NN, F1);

  // ---------------- layer 2: h2 = A2 @ W2 (bf16 WMMA) ----------------
  k_f32_to_bf16_T<<<blks(F1 * F2), 256, 0, stream>>>(W2, W2b, F1, F2);
  k_wmma_gemm_bf16<<<dim3(F2 / BN, (NN + BM - 1) / BM), 256, 0, stream>>>(
      A2, W2b, h2, NN, F1, F2);

  k_attn_scores<<<(NN + 7) / 8, 256, 0, stream>>>(
      h2, asrc2, adst2, als2, ald2, NN, 1, F2);
  k_zero32<<<blks(NN), 256, 0, stream>>>(m2, NN);
  k_zero32<<<blks(NN), 256, 0, stream>>>((unsigned*)den2, NN);
  k_zero32<<<blks(NN * F2), 256, 0, stream>>>((unsigned*)out2, NN * F2);
  k_edge_scores<<<blks(ETOT), 256, 0, stream>>>(
      srcE, dstE, als2, ald2, eb2, m2, ETOT, NE, 1);
  k_edge_expsum<<<blks(ETOT), 256, 0, stream>>>(
      dstE, eb2, m2, den2, ETOT, NE, 1);
  k_edge_aggregate<<<ETOT, 64, 0, stream>>>(
      srcE, dstE, h2, eb2, den2, out2, ETOT, NE, 1, F2);

  // concat=False with H=1 -> mean is identity; just add bias into d_out
  k_add_bias<<<blks(NN * F2), 256, 0, stream>>>(out2, b2, (float*)d_out, NN, F2);
}
